// Decoder_61564061221219
// MI455X (gfx1250) — matrix operand order repaired and checked by reading, not compiled, not run on hardware
//
#include <hip/hip_runtime.h>

#define L_LAYERS 6
#define DMODEL 512
#define NHEAD 8
#define DHEAD 64
#define FFDIM 2048
#define DLAT 128
#define BATCH 16
#define TLEN 512
#define SLEN 512
#define MROWS (BATCH * TLEN) /* 8192 */

typedef __attribute__((ext_vector_type(16))) __bf16 bf16x16;
typedef __attribute__((ext_vector_type(8)))  __bf16 bf16x8;
typedef __attribute__((ext_vector_type(8)))  float  v8f;

__device__ __forceinline__ __bf16 f2bf(float x) {
  union { float f; unsigned u; } v; v.f = x;
  unsigned r = v.u + 0x7FFFu + ((v.u >> 16) & 1u);
  union { unsigned short s; __bf16 b; } o; o.s = (unsigned short)(r >> 16);
  return o.b;
}

__device__ __forceinline__ bf16x16 cat8(bf16x8 lo, bf16x8 hi) {
  bf16x16 r;
  #pragma unroll
  for (int i = 0; i < 8; ++i) { r[i] = lo[i]; r[i + 8] = hi[i]; }
  return r;
}

// 16 contiguous bf16 -> one fragment register set (two b128 loads)
__device__ __forceinline__ bf16x16 ldfrag(const __bf16* p) {
  return cat8(*(const bf16x8*)p, *(const bf16x8*)(p + 8));
}

__device__ __forceinline__ v8f wmma_bf16(bf16x16 a, bf16x16 b, v8f c) {
  return __builtin_amdgcn_wmma_f32_16x16x32_bf16(
      false, a, false, b, (short)0, c, false, false);
}

// CDNA5 async global->LDS copy (16 bytes/lane), tracked by ASYNCcnt.
// LDS byte address = low 32 bits of the flat shared pointer (ISA 10.2:
// LDS aperture addresses truncate to addr[31:0]).
__device__ __forceinline__ void async_copy_b128(__bf16* ldst,
                                                const __bf16* gsrc) {
  unsigned l = (unsigned)(unsigned long long)(uintptr_t)ldst;
  unsigned long long g = (unsigned long long)(uintptr_t)gsrc;
  asm volatile("global_load_async_to_lds_b128 %0, %1, off"
               :: "v"(l), "v"(g)
               : "memory");
}
__device__ __forceinline__ void async_wait0() {
  asm volatile("s_wait_asynccnt 0" ::: "memory");
}

// ---------------------------------------------------------------------------
// Weight convert+transpose: in f32 [K,N] (matrix blockIdx.z) -> out bf16 [N,K]
// ---------------------------------------------------------------------------
__global__ __launch_bounds__(256) void wconvert_kernel(
    const float* __restrict__ in, __bf16* __restrict__ out, int K, int N) {
  __shared__ float tile[32][33];
  const float* src = in + (size_t)blockIdx.z * K * N;
  __bf16* dst = out + (size_t)blockIdx.z * K * N;
  const int k0 = blockIdx.y * 32, n0 = blockIdx.x * 32;
  for (int i = threadIdx.x; i < 1024; i += 256) {
    int r = i >> 5, c = i & 31;
    tile[r][c] = src[(size_t)(k0 + r) * N + n0 + c];
  }
  __syncthreads();
  for (int i = threadIdx.x; i < 1024; i += 256) {
    int n = i >> 5, k = i & 31;
    dst[(size_t)(n0 + n) * K + k0 + k] = f2bf(tile[k][n]);
  }
}

// Elementwise f32 -> bf16 (8 per thread)
__global__ __launch_bounds__(256) void econvert_kernel(
    const float* __restrict__ in, __bf16* __restrict__ out) {
  const size_t i = ((size_t)blockIdx.x * 256 + threadIdx.x) * 8;
  const float4 f0 = *(const float4*)(in + i);
  const float4 f1 = *(const float4*)(in + i + 4);
  bf16x8 r;
  r[0] = f2bf(f0.x); r[1] = f2bf(f0.y); r[2] = f2bf(f0.z); r[3] = f2bf(f0.w);
  r[4] = f2bf(f1.x); r[5] = f2bf(f1.y); r[6] = f2bf(f1.z); r[7] = f2bf(f1.w);
  *(bf16x8*)(out + i) = r;
}

// ---------------------------------------------------------------------------
// GEMM: out[M,N] = op(A[M,K] @ Wt[N,K]^T + bias[N] (+ R[M,N]))
// A, Wt are bf16; accumulation f32.
// ACT: 0 none, 1 lrelu, 2 2*lrelu.  STORE: 0 f32, 1 bf16, 2 bf16 V-transposed.
// Block tile 128x64, BK=32, 128 threads (4 waves), wave = 32 rows x 64 cols.
// W tile staged to LDS via CDNA5 async direct-to-LDS loads (no VGPR
// round-trip); A fragments loaded directly from global (no cross-wave reuse).
// ---------------------------------------------------------------------------
template <int ACT, bool RES, int STORE>
__global__ __launch_bounds__(128) void gemm_kernel(
    const __bf16* __restrict__ A, const __bf16* __restrict__ Wt,
    const float* __restrict__ bias, const float* __restrict__ R,
    void* __restrict__ outp, int M, int N, int K) {
  __shared__ __align__(16) __bf16 sW[64 * 32];  // [n][k]
  const int tid  = threadIdx.x;
  const int wid  = tid >> 5;
  const int lane = tid & 31;
  const int half = lane >> 4;
  const int lrow = lane & 15;
  const int n0 = blockIdx.x * 64;
  const int m0 = blockIdx.y * 128;

  v8f acc[2][4];
  #pragma unroll
  for (int s = 0; s < 2; ++s)
    #pragma unroll
    for (int c = 0; c < 4; ++c) acc[s][c] = (v8f){};

  const __bf16* arow0 = A + (size_t)(m0 + wid * 32 + lrow) * K;
  const __bf16* arow1 = arow0 + (size_t)16 * K;

  for (int k0 = 0; k0 < K; k0 += 32) {
    // stage Wt tile [64][32]: async global->LDS, 16B per lane, 2 chunks/thread
    #pragma unroll
    for (int i = 0; i < 2; ++i) {
      const int idx = tid + i * 128;         // 256 chunks of 8 bf16
      const int n = idx >> 2, ko = (idx & 3) * 8;
      const __bf16* wsrc = Wt + (size_t)(n0 + n) * K + k0 + ko;
      async_copy_b128(sW + n * 32 + ko, wsrc);
      __builtin_prefetch(wsrc + 32, 0, 3);   // next k-tile (speculative)
    }
    async_wait0();
    __syncthreads();
    const bf16x16 a0 = cat8(*(const bf16x8*)(arow0 + k0 + 8 * half),
                            *(const bf16x8*)(arow0 + k0 + 16 + 8 * half));
    const bf16x16 a1 = cat8(*(const bf16x8*)(arow1 + k0 + 8 * half),
                            *(const bf16x8*)(arow1 + k0 + 16 + 8 * half));
    #pragma unroll
    for (int c = 0; c < 4; ++c) {
      const bf16x16 b = cat8(*(const bf16x8*)(sW + (c * 16 + lrow) * 32 + 8 * half),
                             *(const bf16x8*)(sW + (c * 16 + lrow) * 32 + 16 + 8 * half));
      acc[0][c] = wmma_bf16(a0, b, acc[0][c]);
      acc[1][c] = wmma_bf16(a1, b, acc[1][c]);
    }
    __syncthreads();
  }

  #pragma unroll
  for (int s = 0; s < 2; ++s) {
    #pragma unroll
    for (int c = 0; c < 4; ++c) {
      #pragma unroll
      for (int r = 0; r < 8; ++r) {
        const int gm = m0 + wid * 32 + s * 16 + r + 8 * half;
        const int gn = n0 + c * 16 + lrow;
        float v = acc[s][c][r] + bias[gn];
        if (RES) v += R[(size_t)gm * N + gn];
        if (ACT >= 1) v = (v > 0.f) ? v : 0.01f * v;
        if (ACT == 2) v *= 2.f;
        if (STORE == 0) {
          ((float*)outp)[(size_t)gm * N + gn] = v;
        } else if (STORE == 1) {
          ((__bf16*)outp)[(size_t)gm * N + gn] = f2bf(v);
        } else {
          // [B,L,512] element -> per-head transposed [B][H][DH=64][L'=512]
          const int b = gm >> 9, l = gm & 511;
          const int f = l * 512 + gn;  // flat within batch (reshape trick)
          const int h = f >> 15, rem = f & 32767;
          const int lp = rem >> 6, dp = rem & 63;
          ((__bf16*)outp)[(size_t)((b * NHEAD + h) * DHEAD + dp) * 512 + lp] =
              f2bf(v);
        }
      }
    }
  }
}

// ---------------------------------------------------------------------------
// Flash attention, all-bf16 operands. Q,K: [B*H,512,64]; Vt: [B*H,64,512];
// O: [B*H,512,64] bf16. Scale 1/32 (reference bug: score / (DH/2)).
// Grid (qblocks=8, B*H=128), 128 threads; each wave owns 16 q-rows.
// ---------------------------------------------------------------------------
template <bool CAUSAL>
__global__ __launch_bounds__(128) void attention_kernel(
    const __bf16* __restrict__ Q, const __bf16* __restrict__ K,
    const __bf16* __restrict__ Vt, __bf16* __restrict__ O) {
  __shared__ __align__(16) __bf16 sP[4][16 * 32];
  const int bh   = blockIdx.y;
  const int wid  = threadIdx.x >> 5;
  const int lane = threadIdx.x & 31;
  const int half = lane >> 4;
  const int lrow = lane & 15;
  const int q0 = blockIdx.x * 64 + wid * 16;
  const __bf16* Qb = Q + (size_t)(bh * 512 + q0) * DHEAD;
  const __bf16* Kb = K + (size_t)bh * 512 * DHEAD;
  const __bf16* Vb = Vt + (size_t)bh * DHEAD * 512;
  __bf16* myP = sP[wid];

  bf16x16 qa[2];
  #pragma unroll
  for (int j = 0; j < 2; ++j) {
    const __bf16* qr = Qb + lrow * DHEAD + j * 32;
    qa[j] = cat8(*(const bf16x8*)(qr + 8 * half),
                 *(const bf16x8*)(qr + 16 + 8 * half));
  }

  v8f o[4];
  #pragma unroll
  for (int c = 0; c < 4; ++c) o[c] = (v8f){};
  float m_r[8], l_r[8];
  #pragma unroll
  for (int r = 0; r < 8; ++r) { m_r[r] = -1e30f; l_r[r] = 0.f; }

  const int kend = CAUSAL ? (q0 + 16) : SLEN;
  const float sc = 1.0f / 32.0f;

  for (int kb = 0; kb < kend; kb += 32) {
    v8f s0 = (v8f){}, s1 = (v8f){};
    #pragma unroll
    for (int j = 0; j < 2; ++j) {  // d blocks of 32
      const __bf16* k0p = Kb + (size_t)(kb + lrow) * DHEAD + j * 32;
      const __bf16* k1p = Kb + (size_t)(kb + 16 + lrow) * DHEAD + j * 32;
      s0 = wmma_bf16(qa[j],
                     cat8(*(const bf16x8*)(k0p + 8 * half), *(const bf16x8*)(k0p + 16 + 8 * half)),
                     s0);
      s1 = wmma_bf16(qa[j],
                     cat8(*(const bf16x8*)(k1p + 8 * half), *(const bf16x8*)(k1p + 16 + 8 * half)),
                     s1);
    }
    #pragma unroll
    for (int r = 0; r < 8; ++r) {
      const int qrow = q0 + r + 8 * half;
      float a0 = s0[r] * sc;
      float a1 = s1[r] * sc;
      if (CAUSAL) {
        if (kb + lrow > qrow)      a0 = -1e9f;
        if (kb + 16 + lrow > qrow) a1 = -1e9f;
      }
      float loc = fmaxf(a0, a1);
      #pragma unroll
      for (int msk = 8; msk >= 1; msk >>= 1)
        loc = fmaxf(loc, __shfl_xor(loc, msk, 32));
      const float mnew = fmaxf(m_r[r], loc);
      const float corr = __expf(m_r[r] - mnew);
      const float p0 = __expf(a0 - mnew);
      const float p1 = __expf(a1 - mnew);
      float ps = p0 + p1;
      #pragma unroll
      for (int msk = 8; msk >= 1; msk >>= 1) ps += __shfl_xor(ps, msk, 32);
      l_r[r] = l_r[r] * corr + ps;
      m_r[r] = mnew;
      #pragma unroll
      for (int c = 0; c < 4; ++c) o[c][r] *= corr;
      const int prow = r + 8 * half;
      myP[prow * 32 + lrow]      = f2bf(p0);
      myP[prow * 32 + 16 + lrow] = f2bf(p1);
    }
    // P (16x32) A-fragment from wave-private LDS (same-wave DS is in-order)
    const bf16x16 pa = cat8(*(const bf16x8*)(myP + lrow * 32 + 8 * half),
                            *(const bf16x8*)(myP + lrow * 32 + 16 + 8 * half));
    #pragma unroll
    for (int c = 0; c < 4; ++c) {
      const __bf16* vp = Vb + (size_t)(c * 16 + lrow) * 512 + kb;
      o[c] = wmma_bf16(pa, cat8(*(const bf16x8*)(vp + 8 * half), *(const bf16x8*)(vp + 16 + 8 * half)),
                       o[c]);
    }
  }

  #pragma unroll
  for (int r = 0; r < 8; ++r) {
    const float inv = 1.0f / l_r[r];
    const int row = q0 + r + 8 * half;
    #pragma unroll
    for (int c = 0; c < 4; ++c)
      O[(size_t)(bh * 512 + row) * DHEAD + c * 16 + lrow] = f2bf(o[c][r] * inv);
  }
}

// ---------------------------------------------------------------------------
// LayerNorm over D=512; dual store (f32 for residual path, bf16 for WMMA A).
// One wave per row, 8 rows per 256-thread block.
// ---------------------------------------------------------------------------
__global__ __launch_bounds__(256) void layernorm_kernel(
    const float* __restrict__ x, const float* __restrict__ w,
    const float* __restrict__ b, float* __restrict__ y,
    __bf16* __restrict__ ybf) {
  const int wid  = threadIdx.x >> 5;
  const int lane = threadIdx.x & 31;
  const int row = blockIdx.x * 8 + wid;
  const float* xr = x + (size_t)row * DMODEL;
  float v[16];
  float s = 0.f;
  #pragma unroll
  for (int i = 0; i < 16; ++i) { v[i] = xr[lane + i * 32]; s += v[i]; }
  #pragma unroll
  for (int m = 16; m >= 1; m >>= 1) s += __shfl_xor(s, m, 32);
  const float mean = s * (1.f / DMODEL);
  float var = 0.f;
  #pragma unroll
  for (int i = 0; i < 16; ++i) { float d = v[i] - mean; var += d * d; }
  #pragma unroll
  for (int m = 16; m >= 1; m >>= 1) var += __shfl_xor(var, m, 32);
  const float inv = rsqrtf(var * (1.f / DMODEL) + 1e-5f);
  float* yr = y + (size_t)row * DMODEL;
  __bf16* yb = ybf + (size_t)row * DMODEL;
  #pragma unroll
  for (int i = 0; i < 16; ++i) {
    const int col = lane + i * 32;
    const float o = (v[i] - mean) * inv * w[col] + b[col];
    yr[col] = o;
    yb[col] = f2bf(o);
  }
}

// ---------------------------------------------------------------------------
extern "C" void kernel_launch(void* const* d_in, const int* in_sizes, int n_in,
                              void* d_out, int out_size, void* d_ws,
                              size_t ws_size, hipStream_t stream) {
  (void)in_sizes; (void)n_in; (void)out_size; (void)ws_size;
  const float* memory   = (const float*)d_in[0];
  const float* target   = (const float*)d_in[1];
  /* d_in[2] = mask (int32): causality computed analytically */
  const float* expand_w = (const float*)d_in[3];
  const float* expand_b = (const float*)d_in[4];
  const float* ln_w     = (const float*)d_in[5];   // [6,3,512]
  const float* ln_b     = (const float*)d_in[6];
  const float* attn_w   = (const float*)d_in[7];   // [6,2,4,512,512]
  const float* attn_b   = (const float*)d_in[8];   // [6,2,4,512]
  const float* ff_w1    = (const float*)d_in[9];   // [6,512,2048]
  const float* ff_b1    = (const float*)d_in[10];  // [6,2048]
  const float* ff_w2    = (const float*)d_in[11];  // [6,2048,512]
  const float* ff_b2    = (const float*)d_in[12];  // [6,512]

  const size_t NT = (size_t)MROWS * DMODEL;  // 4M elements
  char* base = (char*)d_ws;
  size_t off = 0;
  auto alloc = [&](size_t bytes) -> void* {
    void* p = base + off;
    off = (off + bytes + 255) & ~(size_t)255;
    return p;
  };
  // bf16 pre-transposed weights
  __bf16* wt_attn = (__bf16*)alloc((size_t)48 * 512 * 512 * 2);
  __bf16* wt_ff1  = (__bf16*)alloc((size_t)6 * 2048 * 512 * 2);
  __bf16* wt_ff2  = (__bf16*)alloc((size_t)6 * 512 * 2048 * 2);
  __bf16* wt_exp  = (__bf16*)alloc((size_t)512 * 128 * 2);
  // bf16 activations
  __bf16* mem_b16 = (__bf16*)alloc((size_t)MROWS * DLAT * 2);  // memory input
  __bf16* memb    = (__bf16*)alloc(NT * 2);  // expanded memory
  __bf16* tlnb    = (__bf16*)alloc(NT * 2);
  __bf16* Qbf     = (__bf16*)alloc(NT * 2);
  __bf16* Kbf     = (__bf16*)alloc(NT * 2);
  __bf16* Vbf     = (__bf16*)alloc(NT * 2);  // per-head transposed
  __bf16* aObf    = (__bf16*)alloc(NT * 2);
  __bf16* ffhb    = Qbf;  // 16M bf16; overlaps dead Q/K/V/aO in FF phase
  // f32 state
  float* t   = (float*)alloc(NT * 4);
  float* tln = (float*)alloc(NT * 4);

  dim3 blk(128);
  dim3 g512(DMODEL / 64, MROWS / 128);
  dim3 gff1(FFDIM / 64, MROWS / 128);
  dim3 gattn(TLEN / 64, BATCH * NHEAD);

  // ---- one-time (per launch) weight conversion + transpose ----
  wconvert_kernel<<<dim3(16, 16, 48), 256, 0, stream>>>(attn_w, wt_attn, 512, 512);
  wconvert_kernel<<<dim3(64, 16, 6), 256, 0, stream>>>(ff_w1, wt_ff1, 512, 2048);
  wconvert_kernel<<<dim3(16, 64, 6), 256, 0, stream>>>(ff_w2, wt_ff2, 2048, 512);
  wconvert_kernel<<<dim3(16, 4, 1), 256, 0, stream>>>(expand_w, wt_exp, 128, 512);
  econvert_kernel<<<(MROWS * DLAT) / (256 * 8), 256, 0, stream>>>(memory, mem_b16);

  // mem = leaky_relu(memory @ expand_w + expand_b) -> bf16
  gemm_kernel<1, false, 1><<<g512, blk, 0, stream>>>(
      mem_b16, wt_exp, expand_b, nullptr, memb, MROWS, DMODEL, DLAT);
  hipMemcpyAsync(t, target, NT * sizeof(float), hipMemcpyDeviceToDevice,
                 stream);

  for (int l = 0; l < L_LAYERS; ++l) {
    const __bf16* AW = wt_attn + (size_t)l * 2 * 4 * 512 * 512;
    const float* AB = attn_b + (size_t)l * 2 * 4 * 512;
#define W_(j, p) (AW + (size_t)((j) * 4 + (p)) * 512 * 512)
#define B_(j, p) (AB + (size_t)((j) * 4 + (p)) * 512)
    const float* LW = ln_w + (size_t)l * 3 * 512;
    const float* LB = ln_b + (size_t)l * 3 * 512;

    // ---- self attention ----
    layernorm_kernel<<<MROWS / 8, 256, 0, stream>>>(t, LW, LB, tln, tlnb);
    gemm_kernel<0, false, 1><<<g512, blk, 0, stream>>>(
        tlnb, W_(0, 0), B_(0, 0), nullptr, Qbf, MROWS, 512, 512);
    gemm_kernel<0, false, 1><<<g512, blk, 0, stream>>>(
        tlnb, W_(0, 1), B_(0, 1), nullptr, Kbf, MROWS, 512, 512);
    gemm_kernel<0, false, 2><<<g512, blk, 0, stream>>>(
        tlnb, W_(0, 2), B_(0, 2), nullptr, Vbf, MROWS, 512, 512);
    attention_kernel<true><<<gattn, blk, 0, stream>>>(Qbf, Kbf, Vbf, aObf);
    gemm_kernel<0, true, 0><<<g512, blk, 0, stream>>>(
        aObf, W_(0, 3), B_(0, 3), tln, t, MROWS, 512, 512);

    // ---- cross attention ----
    layernorm_kernel<<<MROWS / 8, 256, 0, stream>>>(t, LW + 512, LB + 512, tln,
                                                    tlnb);
    gemm_kernel<0, false, 1><<<g512, blk, 0, stream>>>(
        tlnb, W_(1, 0), B_(1, 0), nullptr, Qbf, MROWS, 512, 512);
    gemm_kernel<0, false, 1><<<g512, blk, 0, stream>>>(
        memb, W_(1, 1), B_(1, 1), nullptr, Kbf, MROWS, 512, 512);
    gemm_kernel<0, false, 2><<<g512, blk, 0, stream>>>(
        memb, W_(1, 2), B_(1, 2), nullptr, Vbf, MROWS, 512, 512);
    attention_kernel<false><<<gattn, blk, 0, stream>>>(Qbf, Kbf, Vbf, aObf);
    gemm_kernel<0, true, 0><<<g512, blk, 0, stream>>>(
        aObf, W_(1, 3), B_(1, 3), tln, t, MROWS, 512, 512);

    // ---- feed forward: t = 2*lrelu(lrelu(LN(t)@W1+b1)@W2+b2) ----
    layernorm_kernel<<<MROWS / 8, 256, 0, stream>>>(t, LW + 1024, LB + 1024,
                                                    tln, tlnb);
    gemm_kernel<1, false, 1><<<gff1, blk, 0, stream>>>(
        tlnb, wt_ff1 + (size_t)l * 2048 * 512, ff_b1 + (size_t)l * 2048,
        nullptr, ffhb, MROWS, FFDIM, 512);
    gemm_kernel<2, false, 0><<<g512, blk, 0, stream>>>(
        ffhb, wt_ff2 + (size_t)l * 512 * 2048, ff_b2 + (size_t)l * 512, nullptr,
        t, MROWS, 512, FFDIM);
#undef W_
#undef B_
  }

  hipMemcpyAsync(d_out, t, NT * sizeof(float), hipMemcpyDeviceToDevice,
                 stream);
}
